// BiLSTM_CRF_45801531244684
// MI455X (gfx1250) — matrix operand order repaired and checked by reading, not compiled, not run on hardware
//
#include <hip/hip_runtime.h>
#include <hip/hip_bf16.h>
#include <math.h>

// ---------------------------------------------------------------------------
// BiLSTM-CRF forward NLL on gfx1250 (wave32, WMMA bf16 -> f32, TDM prefetch)
// B=128, S=512, D=128, H=128, G=4H=512, T=20
// ---------------------------------------------------------------------------

#define NB   128
#define NS   512
#define ND   128
#define NH   128
#define NG   512      // 4*H
#define NT   20
#define BS   (NB * NS)   // 65536

typedef __attribute__((ext_vector_type(16))) __bf16 v16bf;
typedef __attribute__((ext_vector_type(8)))  float  v8f;
typedef __attribute__((ext_vector_type(4)))  unsigned int u32x4;
typedef __attribute__((ext_vector_type(8)))  int i32x8;
typedef __attribute__((ext_vector_type(4)))  int i32x4;

// ---- WMMA fragment loaders (ISA 7.12.2 layouts, wave32) -------------------

// A matrix 16x32 bf16, row-major with row stride lda. a00 = tile origin.
static __device__ __forceinline__ v16bf load_a_tile(const __bf16* a00, int lda, int lane) {
    int m  = lane & 15;
    int kh = (lane & 16) ? 8 : 0;
    v16bf r;
#pragma unroll
    for (int v = 0; v < 8; ++v) {
        int kk = ((v & 4) ? 16 : 0) + kh + 2 * (v & 3);
        const __bf16* p = a00 + m * lda + kk;
        r[2 * v]     = p[0];
        r[2 * v + 1] = p[1];
    }
    return r;
}

// B matrix 32x16 bf16 from W stored (N x K) row-major (B = W^T).
static __device__ __forceinline__ v16bf load_b_tile(const __bf16* w00, int ldw, int lane) {
    int n  = lane & 15;
    int kh = (lane & 16) ? 8 : 0;
    v16bf r;
#pragma unroll
    for (int v = 0; v < 8; ++v) {
        int kk = ((v & 4) ? 16 : 0) + kh + 2 * (v & 3);
        const __bf16* p = w00 + n * ldw + kk;
        r[2 * v]     = p[0];
        r[2 * v + 1] = p[1];
    }
    return r;
}

// ---- TDM tile load: strided 2D tile (16 rows x NG bf16, row stride NS*NG) --
#if __has_builtin(__builtin_amdgcn_tensor_load_to_lds) && \
    __has_builtin(__builtin_amdgcn_s_wait_tensorcnt)
#define HAVE_TDM 1
#endif

#ifdef HAVE_TDM
static __device__ __forceinline__ void tdm_load_xp_tile(const __bf16* gsrc,
                                                        unsigned lds_off) {
    unsigned long long ga = (unsigned long long)(size_t)gsrc;
    u32x4 g0;
    g0[0] = 1u;                                          // count=1, user D#
    g0[1] = lds_off;                                     // lds_addr (bytes)
    g0[2] = (unsigned)(ga & 0xffffffffu);                // global_addr lo
    g0[3] = (unsigned)((ga >> 32) & 0x01ffffffu)         // global_addr hi (57b)
          | (2u << 30);                                  // type = 2 ("image")
    i32x8 g1;
    g1[0] = (1 << 16);                 // wg_mask=0, data_size=1 (2 bytes)
    g1[1] = (int)((unsigned)NG << 16); // tensor_dim0[15:0] at bit 48
    g1[2] = (int)(16u << 16);          // tensor_dim0 hi=0 | tensor_dim1=16
    g1[3] = (int)((unsigned)NG << 16); // tensor_dim1 hi=0 | tile_dim0=NG
    g1[4] = 16;                        // tile_dim1=16, tile_dim2=0
    g1[5] = (int)(NS * NG);            // tensor_dim0_stride lo32 (elements)
    g1[6] = 0;                         // stride hi | dim1_stride lo
    g1[7] = 0;
    i32x4 z4 = {0, 0, 0, 0};           // groups 2/3 unused (2D tensor)
    i32x8 z8 = {0, 0, 0, 0, 0, 0, 0, 0};
    __builtin_amdgcn_tensor_load_to_lds(g0, g1, z4, z4, z8, 0);
}
#endif

// ---- small utility kernels ------------------------------------------------

__global__ void zero_out_kernel(float* o) { o[0] = 0.0f; }

__global__ void cvt_f32_bf16(const float* __restrict__ s, __bf16* __restrict__ d, int n) {
    int i = blockIdx.x * 256 + threadIdx.x;
    if (i < n) d[i] = (__bf16)s[i];
}

// Pad w_out (20 x 256) to (32 x 256) bf16, zero rows >= 20.
__global__ void pad_wout_bf16(const float* __restrict__ s, __bf16* __restrict__ d) {
    int i = blockIdx.x * 256 + threadIdx.x;
    if (i >= 32 * 256) return;
    int row = i >> 8, col = i & 255;
    d[i] = (row < NT) ? (__bf16)s[row * 256 + col] : (__bf16)0.0f;
}

// ---- xp GEMM: C(MxN,bf16) = A(MxK,bf16) @ W(NxK,bf16)^T + bias ------------
// Wave computes a 16x64 strip: A fragment reused across 4 WMMAs.
__global__ __launch_bounds__(256)
void gemm4_bf16_wmma(const __bf16* __restrict__ A, const __bf16* __restrict__ W,
                     const float* __restrict__ bias, __bf16* __restrict__ C,
                     int M, int N, int K, int lda, int ldw, int ldc) {
    int lane = threadIdx.x & 31;
    int wave = threadIdx.x >> 5;
    long wid = (long)blockIdx.x * 8 + wave;
    int n64 = N >> 6;
    long tm = wid / n64;
    int  tn = (int)(wid % n64);
    if (tm >= (M >> 4)) return;
    int m0 = (int)tm << 4;
    int n0 = tn << 6;

    const __bf16* a_row = A + (size_t)m0 * lda;
    v8f acc[4] = {};
    for (int k0 = 0; k0 < K; k0 += 32) {
        v16bf a = load_a_tile(a_row + k0, lda, lane);
#pragma unroll
        for (int t = 0; t < 4; ++t) {
            v16bf b = load_b_tile(W + (size_t)(n0 + t * 16) * ldw + k0, ldw, lane);
            acc[t] = __builtin_amdgcn_wmma_f32_16x16x32_bf16(
                false, a, false, b, (short)0, acc[t], false, false);
        }
    }
    int n  = lane & 15;
    int mh = (lane & 16) ? 8 : 0;
#pragma unroll
    for (int t = 0; t < 4; ++t) {
        int gn = n0 + t * 16 + n;
        float bv = bias[gn];
#pragma unroll
        for (int v = 0; v < 8; ++v)
            C[(size_t)(m0 + v + mh) * ldc + gn] = (__bf16)(acc[t][v] + bv);
    }
}

// ---- emissions GEMM: C(Mx20,f32) = A(Mx256,bf16) @ W(32x256,bf16)^T + b ---
__global__ __launch_bounds__(256)
void gemm_emis_wmma(const __bf16* __restrict__ A, const __bf16* __restrict__ W,
                    const float* __restrict__ bias, float* __restrict__ C,
                    int M, int N, int K, int lda, int ldw, int ldc, int Nvalid) {
    int lane = threadIdx.x & 31;
    int wave = threadIdx.x >> 5;
    long tile = (long)blockIdx.x * 8 + wave;
    int tilesN = N >> 4;
    long tm = tile / tilesN;
    int  tn = (int)(tile % tilesN);
    if (tm >= (M >> 4)) return;
    int m0 = (int)tm << 4;
    int n0 = tn << 4;

    const __bf16* a_row = A + (size_t)m0 * lda;
    const __bf16* w_row = W + (size_t)n0 * ldw;
    v8f acc = {};
    for (int k0 = 0; k0 < K; k0 += 32) {
        v16bf a = load_a_tile(a_row + k0, lda, lane);
        v16bf b = load_b_tile(w_row + k0, ldw, lane);
        acc = __builtin_amdgcn_wmma_f32_16x16x32_bf16(false, a, false, b,
                                                      (short)0, acc, false, false);
    }
    int n  = lane & 15;
    int mh = (lane & 16) ? 8 : 0;
    int gn = n0 + n;
    if (gn < Nvalid) {
        float bv = bias[gn];
#pragma unroll
        for (int v = 0; v < 8; ++v)
            C[(size_t)(m0 + v + mh) * ldc + gn] = acc[v] + bv;
    }
}

// ---- persistent LSTM recurrence kernel ------------------------------------
// grid.x = 8 batch tiles of M=16. 8 waves; wave w owns gate cols [64w,64w+64).
// Pipeline per step: WMMA on h_lds (xp tile for this step already in flight
// from the TDM issue at the end of the previous step) -> gates to LDS ->
// s_wait_tensorcnt + barrier -> elementwise update (gates + xp from LDS) ->
// barrier -> TDM issue for next step's xp tile.
__global__ __launch_bounds__(256)
void lstm_kernel(const __bf16* __restrict__ xp, const __bf16* __restrict__ whh,
                 __bf16* __restrict__ hs_cat, int dir_off, int reverse) {
    __shared__ __bf16 h_lds[16 * NH];      // 4 KB
    __shared__ float  g_lds[16 * NG];      // 32 KB
    __shared__ __bf16 xp_lds[16 * NG];     // 16 KB staging for TDM

    int tid  = threadIdx.x;
    int lane = tid & 31;
    int wave = tid >> 5;
    int m0   = blockIdx.x * 16;
    int n_base = wave * 64;

    // xp element (m, t, n) lives at ((m0+m)*NS + t)*NG + n
    const __bf16* xp_base = xp + (size_t)m0 * NS * NG;
#ifdef HAVE_TDM
    unsigned xp_lds_off = (unsigned)(size_t)&xp_lds[0];
#endif

    float c_reg[8];
#pragma unroll
    for (int e = 0; e < 8; ++e) {
        c_reg[e] = 0.0f;
        h_lds[tid * 8 + e] = (__bf16)0.0f;
    }

    // prefetch xp tile for the first step
    {
        int ts0 = reverse ? (NS - 1) : 0;
#ifdef HAVE_TDM
        if (wave == 0)
            tdm_load_xp_tile(xp_base + (size_t)ts0 * NG, xp_lds_off);
#else
        for (int i = tid; i < 16 * NG / 8; i += 256) {
            int m = i >> 6, c = i & 63;
            *(uint4*)(xp_lds + m * NG + c * 8) =
                *(const uint4*)(xp_base + (size_t)m * NS * NG + (size_t)ts0 * NG + c * 8);
        }
#endif
    }
    __syncthreads();

    for (int step = 0; step < NS; ++step) {
        int ts = reverse ? (NS - 1 - step) : step;

        // h(16x128,bf16) @ w_hh^T slice: 4 K-steps x 4 N-tiles of WMMA
        v8f acc[4] = {};
#pragma unroll
        for (int kt = 0; kt < 4; ++kt) {
            v16bf a = load_a_tile(h_lds + kt * 32, NH, lane);
#pragma unroll
            for (int tn = 0; tn < 4; ++tn) {
                v16bf b = load_b_tile(whh + (size_t)(n_base + tn * 16) * NH + kt * 32,
                                      NH, lane);
                acc[tn] = __builtin_amdgcn_wmma_f32_16x16x32_bf16(
                    false, a, false, b, (short)0, acc[tn], false, false);
            }
        }

        // scatter raw recurrent gates to LDS
#pragma unroll
        for (int tn = 0; tn < 4; ++tn) {
            int n = n_base + tn * 16 + (lane & 15);
#pragma unroll
            for (int v = 0; v < 8; ++v) {
                int m = v + ((lane & 16) ? 8 : 0);
                g_lds[m * NG + n] = acc[tn][v];
            }
        }

#ifdef HAVE_TDM
        if (wave == 0) __builtin_amdgcn_s_wait_tensorcnt(0);
#endif
        __syncthreads();

        // elementwise LSTM cell update; thread owns (m, j0..j0+7)
#pragma unroll
        for (int e = 0; e < 8; ++e) {
            int idx = tid * 8 + e;
            int m = idx >> 7;
            int j = idx & 127;
            float ig = g_lds[m * NG + j]          + (float)xp_lds[m * NG + j];
            float fg = g_lds[m * NG + NH + j]     + (float)xp_lds[m * NG + NH + j];
            float gg = g_lds[m * NG + 2 * NH + j] + (float)xp_lds[m * NG + 2 * NH + j];
            float og = g_lds[m * NG + 3 * NH + j] + (float)xp_lds[m * NG + 3 * NH + j];
            float is = 1.0f / (1.0f + __expf(-ig));
            float fs = 1.0f / (1.0f + __expf(-fg));
            float os = 1.0f / (1.0f + __expf(-og));
            float c  = fs * c_reg[e] + is * tanhf(gg);
            c_reg[e] = c;
            float h  = os * tanhf(c);
            __bf16 hb = (__bf16)h;
            h_lds[idx] = hb;
            hs_cat[((size_t)(m0 + m) * NS + ts) * (2 * NH) + dir_off + j] = hb;
        }
        __syncthreads();

        // issue prefetch of next step's xp tile (overlaps next WMMA phase)
        if (step + 1 < NS) {
            int tn_s = reverse ? (NS - 2 - step) : (step + 1);
#ifdef HAVE_TDM
            if (wave == 0)
                tdm_load_xp_tile(xp_base + (size_t)tn_s * NG, xp_lds_off);
#else
            for (int i = tid; i < 16 * NG / 8; i += 256) {
                int m = i >> 6, c = i & 63;
                *(uint4*)(xp_lds + m * NG + c * 8) =
                    *(const uint4*)(xp_base + (size_t)m * NS * NG +
                                    (size_t)tn_s * NG + c * 8);
            }
#endif
        }
    }
}

// ---- CRF NLL: numerator + forward algorithm, one wave per batch element ---
__global__ __launch_bounds__(32)
void crf_kernel(const float* __restrict__ em, const int* __restrict__ tags,
                const unsigned char* __restrict__ mask,
                const float* __restrict__ start_t, const float* __restrict__ end_t,
                const float* __restrict__ trans, float* __restrict__ out) {
    __shared__ float tr[NT * NT];
    __shared__ float sc[NT];
    __shared__ float red[NT];

    int lane = threadIdx.x;
    int b = blockIdx.x;
    for (int i = lane; i < NT * NT; i += 32) tr[i] = trans[i];
    __syncthreads();

    const float* emb = em + (size_t)b * NS * NT;
    const int* tg = tags + (size_t)b * NS;
    const unsigned char* mk = mask + (size_t)b * NS;

    float num = 0.0f;
    if (lane == 0) {
        int t0 = tg[0];
        num = start_t[t0] + emb[t0];
        int cnt = mk[0] ? 1 : 0;
        for (int s = 1; s < NS; ++s) {
            if (mk[s]) {
                num += tr[tg[s - 1] * NT + tg[s]] + emb[s * NT + tg[s]];
                ++cnt;
            }
        }
        num += end_t[tg[cnt - 1]];
    }

    float score = 0.0f;
    if (lane < NT) score = start_t[lane] + emb[lane];
    for (int s = 1; s < NS; ++s) {
        if (lane < NT) sc[lane] = score;
        __syncthreads();
        float nxt = score;
        if (lane < NT && mk[s]) {
            float mx = -1e30f;
            for (int i = 0; i < NT; ++i)
                mx = fmaxf(mx, sc[i] + tr[i * NT + lane]);
            float sum = 0.0f;
            for (int i = 0; i < NT; ++i)
                sum += __expf(sc[i] + tr[i * NT + lane] - mx);
            nxt = mx + __logf(sum) + emb[s * NT + lane];
        }
        __syncthreads();
        score = nxt;
    }

    if (lane < NT) red[lane] = score + end_t[lane];
    __syncthreads();
    if (lane == 0) {
        float mx = -1e30f;
        for (int i = 0; i < NT; ++i) mx = fmaxf(mx, red[i]);
        float sum = 0.0f;
        for (int i = 0; i < NT; ++i) sum += __expf(red[i] - mx);
        float den = mx + __logf(sum);
        atomicAdd(out, -(num - den) * (1.0f / (float)NB));
    }
}

// ---------------------------------------------------------------------------

extern "C" void kernel_launch(void* const* d_in, const int* in_sizes, int n_in,
                              void* d_out, int out_size, void* d_ws, size_t ws_size,
                              hipStream_t stream) {
    const float* x        = (const float*)d_in[0];
    const int*   tags     = (const int*)d_in[1];
    const unsigned char* mask = (const unsigned char*)d_in[2];
    const float* w_ih_f   = (const float*)d_in[3];
    const float* w_hh_f   = (const float*)d_in[4];
    const float* b_f      = (const float*)d_in[5];
    const float* w_ih_b   = (const float*)d_in[6];
    const float* w_hh_b   = (const float*)d_in[7];
    const float* b_b      = (const float*)d_in[8];
    const float* w_out    = (const float*)d_in[9];
    const float* b_out    = (const float*)d_in[10];
    const float* start_t  = (const float*)d_in[11];
    const float* end_t    = (const float*)d_in[12];
    const float* trans    = (const float*)d_in[13];
    float* out = (float*)d_out;

    // workspace layout
    char* ws = (char*)d_ws;
    size_t off = 0;
    __bf16* xbf    = (__bf16*)(ws + off); off += (size_t)BS * ND * 2;        // 16 MB
    __bf16* wihf   = (__bf16*)(ws + off); off += (size_t)NG * ND * 2;
    __bf16* wihb   = (__bf16*)(ws + off); off += (size_t)NG * ND * 2;
    __bf16* whhf   = (__bf16*)(ws + off); off += (size_t)NG * NH * 2;
    __bf16* whhb   = (__bf16*)(ws + off); off += (size_t)NG * NH * 2;
    __bf16* woutbf = (__bf16*)(ws + off); off += (size_t)32 * 256 * 2;
    __bf16* xp_f   = (__bf16*)(ws + off); off += (size_t)BS * NG * 2;        // 64 MB
    __bf16* xp_b   = (__bf16*)(ws + off); off += (size_t)BS * NG * 2;        // 64 MB
    __bf16* hs_cat = (__bf16*)(ws + off); off += (size_t)BS * 2 * NH * 2;    // 32 MB
    float*  em     = (float*)(ws + off);  off += (size_t)BS * NT * 4;        // 5 MB

    zero_out_kernel<<<1, 1, 0, stream>>>(out);

    // f32 -> bf16 conversions
    {
        int n = BS * ND;
        cvt_f32_bf16<<<(n + 255) / 256, 256, 0, stream>>>(x, xbf, n);
        n = NG * ND;
        cvt_f32_bf16<<<(n + 255) / 256, 256, 0, stream>>>(w_ih_f, wihf, n);
        cvt_f32_bf16<<<(n + 255) / 256, 256, 0, stream>>>(w_ih_b, wihb, n);
        cvt_f32_bf16<<<(n + 255) / 256, 256, 0, stream>>>(w_hh_f, whhf, n);
        cvt_f32_bf16<<<(n + 255) / 256, 256, 0, stream>>>(w_hh_b, whhb, n);
        pad_wout_bf16<<<(32 * 256 + 255) / 256, 256, 0, stream>>>(w_out, woutbf);
    }

    // input projections: xp = x @ w_ih^T + b   (65536 x 512, K=128), bf16 out
    {
        int strips = (BS / 16) * (NG / 64);          // 32768 waves
        int blocks = strips / 8;                     // 4096
        gemm4_bf16_wmma<<<blocks, 256, 0, stream>>>(xbf, wihf, b_f, xp_f,
                                                    BS, NG, ND, ND, ND, NG);
        gemm4_bf16_wmma<<<blocks, 256, 0, stream>>>(xbf, wihb, b_b, xp_b,
                                                    BS, NG, ND, ND, ND, NG);
    }

    // recurrences (8 batch tiles per direction)
    lstm_kernel<<<8, 256, 0, stream>>>(xp_f, whhf, hs_cat, 0,  0);
    lstm_kernel<<<8, 256, 0, stream>>>(xp_b, whhb, hs_cat, NH, 1);

    // emissions: (65536 x 256) @ w_out^T -> (65536 x 20), N padded to 32
    {
        int tiles = (BS / 16) * 2;                   // 8192
        int blocks = tiles / 8;                      // 1024
        gemm_emis_wmma<<<blocks, 256, 0, stream>>>(hs_cat, woutbf, b_out, em,
                                                   BS, 32, 2 * NH, 2 * NH, 2 * NH,
                                                   NT, NT);
    }

    // CRF negative log-likelihood
    crf_kernel<<<NB, 32, 0, stream>>>(em, tags, mask, start_t, end_t, trans, out);

    (void)in_sizes; (void)n_in; (void)out_size; (void)ws_size;
}
